// BranchedNetwork_26740466385308
// MI455X (gfx1250) — matrix operand order repaired and checked by reading, not compiled, not run on hardware
//
#include <hip/hip_runtime.h>
#include <hip/hip_bf16.h>
#include <math.h>

// ---------------------------------------------------------------------------
// BranchedNetwork (MoE head) for MI455X / gfx1250, wave32 + WMMA f16->f32.
//
// Only out[command[i], i, 0] is consumed by the reference, so we counting-sort
// rows by command and run ONE expert per row (4x fewer FLOPs than reference),
// and fuse the 512-dot with W2[:, :, 0] + bias + relu + sigmoid/clamp into the
// GEMM epilogue so the [B,512] hidden tensor never hits memory.
//
// Dominant GEMM: 16384 x 1024 x 512 (17.2 GFLOP) -> v_wmma_f32_16x16x32_f16.
// ---------------------------------------------------------------------------

#define B_ROWS 16384
#define EMB    512
#define K_DIM  1024
#define NCMD   4

#define MTILE      64
#define KBLK       32
#define PERM_LEN   (B_ROWS + NCMD * MTILE)        // padded perm capacity
#define MAX_TILES  ((B_ROWS + NCMD * (MTILE - 1) + MTILE - 1) / MTILE)  // 260

// LDS padded strides (f16 elements); keep 16B alignment of WMMA fragments
#define SA_STRIDE 40   // 32 + 8  (row*80B, 16B aligned; spread banks)
#define SB_STRIDE 40   // 32 + 8  (row*80B, 16B aligned; conflict-free cols)

typedef __attribute__((ext_vector_type(16))) _Float16 v16h;
typedef __attribute__((ext_vector_type(8)))  float    v8f;

union ABReg { v16h v; uint4 q[2]; };

// ---------------------------------------------------------------------------
// Prep kernels
// ---------------------------------------------------------------------------

// W1t[e][n][k] = (f16) W1[e][k][n]   (4 x 512 x 1024)
__global__ void k_w1_transpose(const float* __restrict__ W1,
                               _Float16* __restrict__ W1t) {
    int idx = blockIdx.x * blockDim.x + threadIdx.x;   // 0 .. 4*512*1024-1
    int k = idx & (K_DIM - 1);
    int n = (idx >> 10) & (EMB - 1);
    int e = idx >> 19;
    W1t[idx] = (_Float16)W1[((size_t)e * K_DIM + k) * EMB + n];
}

// w2col[e][h] = W2[e][h][0]
__global__ void k_w2_col0(const float* __restrict__ W2, float* __restrict__ w2col) {
    int idx = blockIdx.x * blockDim.x + threadIdx.x;   // 0..2047
    int e = idx >> 9, h = idx & (EMB - 1);
    w2col[idx] = W2[((size_t)e * EMB + h) * 2 + 0];
}

// meta[0..3]=counts  meta[4..7]=cursors  meta[8..12]=padded segment starts
__global__ void k_sort_init(int* __restrict__ perm, int* __restrict__ meta) {
    int idx = blockIdx.x * blockDim.x + threadIdx.x;
    if (idx < PERM_LEN) perm[idx] = -1;
    if (idx < 16) meta[idx] = 0;
}

__global__ void k_sort_hist(const int* __restrict__ cmd, int* __restrict__ meta) {
    int i = blockIdx.x * blockDim.x + threadIdx.x;
    if (i < B_ROWS) atomicAdd(&meta[cmd[i] & 3], 1);
}

__global__ void k_sort_seg(int* __restrict__ meta) {
    if (threadIdx.x == 0 && blockIdx.x == 0) {
        int s = 0;
        for (int e = 0; e < NCMD; ++e) {
            meta[8 + e] = s;
            s += (meta[e] + (MTILE - 1)) & ~(MTILE - 1);   // pad to M tile
        }
        meta[12] = s;
    }
}

__global__ void k_sort_scatter(const int* __restrict__ cmd, int* __restrict__ meta,
                               int* __restrict__ perm) {
    int i = blockIdx.x * blockDim.x + threadIdx.x;
    if (i < B_ROWS) {
        int e = cmd[i] & 3;
        int pos = meta[8 + e] + atomicAdd(&meta[4 + e], 1);
        perm[pos] = i;
    }
}

// ---------------------------------------------------------------------------
// Main fused kernel: one 64-row tile of one expert, full N=512 per block.
// 8 waves = 4 (M slots of 16 rows) x 2 (N halves of 256 cols).
// ---------------------------------------------------------------------------
__global__ __launch_bounds__(256)
void k_moe_gemm(const float* __restrict__ img,
                const float* __restrict__ meas,
                const float* __restrict__ Wm,
                const float* __restrict__ bmv,
                const float* __restrict__ b1,
                const float* __restrict__ b2,
                const _Float16* __restrict__ W1t,
                const float* __restrict__ w2col,
                const int* __restrict__ perm,
                const int* __restrict__ meta,
                float* __restrict__ out) {
    __shared__ __align__(16) _Float16 sA[MTILE * SA_STRIDE];   // 5.0 KB
    __shared__ __align__(16) _Float16 sB[EMB * SB_STRIDE];     // 40.0 KB
    __shared__ float sB1[EMB];
    __shared__ float sW2[EMB];
    __shared__ float sPart[MTILE * 2];
    __shared__ int   sPerm[MTILE];

    const int tid   = threadIdx.x;
    const int tile0 = blockIdx.x * MTILE;
    if (tile0 >= meta[12]) return;                 // beyond padded total (uniform)

    int e = 0;                                     // expert for this tile
    if (tile0 >= meta[9])  e = 1;
    if (tile0 >= meta[10]) e = 2;
    if (tile0 >= meta[11]) e = 3;

    for (int i = tid; i < EMB; i += 256) {
        sB1[i] = b1[e * EMB + i];
        sW2[i] = w2col[e * EMB + i];
    }
    if (tid < MTILE) sPerm[tid] = perm[tile0 + tid];
    __syncthreads();

    const int lane  = tid & 31;
    const int wave  = tid >> 5;
    const int mslot = wave & 3;     // which 16-row group
    const int nhalf = wave >> 2;    // which 256-col half
    const int cloc  = lane & 15;
    const int hi    = lane >> 4;    // 0: lanes 0-15, 1: lanes 16-31

    const v8f vzero = {0.f, 0.f, 0.f, 0.f, 0.f, 0.f, 0.f, 0.f};
    v8f acc[16];
#pragma unroll
    for (int t = 0; t < 16; ++t) acc[t] = vzero;

    const _Float16* W1e = W1t + (size_t)e * EMB * K_DIM;

    // A-loader assignment: 64 rows x 32 k; thread -> (row, 8 k-elements)
    const int lr   = tid >> 2;          // 0..63
    const int lq   = tid & 3;           // 0..3  -> 8 elements
    const int grow = sPerm[lr];         // global row or -1 (padding)
    // B-loader assignment: 2 rows of 32 f16 per thread
    const int bn0  = tid * 2;

    for (int kb = 0; kb < K_DIM; kb += KBLK) {
        __syncthreads();   // previous compute done before LDS overwrite

        // ---- stage A tile (64 x 32 f16), measurement half computed on the fly
        {
            _Float16* dst = &sA[lr * SA_STRIDE + lq * 8];
            const int k0g = kb + lq * 8;
            if (grow < 0) {
#pragma unroll
                for (int i = 0; i < 8; ++i) dst[i] = (_Float16)0.f;
            } else if (kb < EMB) {                       // image-embedding half
                const float* src = img + (size_t)grow * EMB + k0g;
#pragma unroll
                for (int i = 0; i < 8; ++i) dst[i] = (_Float16)src[i];
            } else {                                     // meas @ W_meas + b_meas
                const float* mrow = meas + (size_t)grow * 8;
                float m0 = mrow[0], m1 = mrow[1], m2 = mrow[2], m3 = mrow[3];
                float m4 = mrow[4], m5 = mrow[5], m6 = mrow[6], m7 = mrow[7];
                const int j0 = k0g - EMB;
#pragma unroll
                for (int i = 0; i < 8; ++i) {
                    const int j = j0 + i;
                    float v = bmv[j];
                    v = fmaf(m0, Wm[0 * EMB + j], v);
                    v = fmaf(m1, Wm[1 * EMB + j], v);
                    v = fmaf(m2, Wm[2 * EMB + j], v);
                    v = fmaf(m3, Wm[3 * EMB + j], v);
                    v = fmaf(m4, Wm[4 * EMB + j], v);
                    v = fmaf(m5, Wm[5 * EMB + j], v);
                    v = fmaf(m6, Wm[6 * EMB + j], v);
                    v = fmaf(m7, Wm[7 * EMB + j], v);
                    dst[i] = (_Float16)v;
                }
            }
        }

        // ---- stage B tile (512 x 32 f16) from pre-transposed W1t (contig K)
        {
#pragma unroll
            for (int rr = 0; rr < 2; ++rr) {
                const int n = bn0 + rr;
                const uint4* src = (const uint4*)(W1e + (size_t)n * K_DIM + kb);
                uint4* dst = (uint4*)(&sB[n * SB_STRIDE]);
#pragma unroll
                for (int q = 0; q < 4; ++q) dst[q] = src[q];
            }
            if (kb + KBLK < K_DIM)                       // warm L2/L0 for next tile
                __builtin_prefetch(W1e + (size_t)bn0 * K_DIM + kb + KBLK, 0, 1);
        }
        __syncthreads();

        // ---- compute: 16 WMMAs per wave per K-block
        // A 16x32 f16 lane layout: lanes 0-15: K 0..7,16..23 ; lanes 16-31: +8
        ABReg a;
        {
            const int arow = mslot * 16 + cloc;
            const int ako  = hi * 8;
            a.q[0] = *(const uint4*)(&sA[arow * SA_STRIDE + ako]);
            a.q[1] = *(const uint4*)(&sA[arow * SA_STRIDE + ako + 16]);
        }
        // B 32x16 f16 lane layout: lanes 0-15: K 0..7,16..23 ; lanes 16-31: +8
#pragma unroll
        for (int t = 0; t < 16; ++t) {
            ABReg breg;
            const int bcol = nhalf * 256 + t * 16 + cloc;
            const int bko  = hi * 8;
            breg.q[0] = *(const uint4*)(&sB[bcol * SB_STRIDE + bko]);
            breg.q[1] = *(const uint4*)(&sB[bcol * SB_STRIDE + bko + 16]);
            acc[t] = __builtin_amdgcn_wmma_f32_16x16x32_f16(
                false, a.v, false, breg.v, (short)0, acc[t], false, false);
        }
    }

    // ---- epilogue: +b1, relu, dot with W2[:,0]; per-wave partial over 256 cols
    float s[8];
#pragma unroll
    for (int r = 0; r < 8; ++r) s[r] = 0.f;
#pragma unroll
    for (int t = 0; t < 16; ++t) {
        const int col  = nhalf * 256 + t * 16 + cloc;
        const float bv = sB1[col];
        const float wv = sW2[col];
#pragma unroll
        for (int r = 0; r < 8; ++r) {
            float h = acc[t][r] + bv;
            h = h > 0.f ? h : 0.f;
            s[r] = fmaf(h, wv, s[r]);
        }
    }
    // reduce across the 16 lanes of each half (C-matrix cols live per-lane)
#pragma unroll
    for (int off = 1; off < 16; off <<= 1) {
#pragma unroll
        for (int r = 0; r < 8; ++r) s[r] += __shfl_xor(s[r], off, 32);
    }
    if (cloc == 0) {
#pragma unroll
        for (int r = 0; r < 8; ++r) {
            const int row = mslot * 16 + hi * 8 + r;   // f32 C/D layout: M = hi*8 + r
            sPart[row * 2 + nhalf] = s[r];
        }
    }
    __syncthreads();

    if (tid < MTILE) {
        const float p = sPart[tid * 2] + sPart[tid * 2 + 1] + b2[e * 2];
        const int g = sPerm[tid];
        if (g >= 0) {
            out[g]          = 50.0f / (1.0f + __expf(-p));        // sigmoid*50
            out[B_ROWS + g] = fminf(1.0f, fmaxf(-1.0f, p));       // hardtanh
        }
    }
}

// ---------------------------------------------------------------------------
// Launch
// ---------------------------------------------------------------------------
extern "C" void kernel_launch(void* const* d_in, const int* in_sizes, int n_in,
                              void* d_out, int out_size, void* d_ws, size_t ws_size,
                              hipStream_t stream) {
    const float* img  = (const float*)d_in[0];   // [B, 512]
    const float* meas = (const float*)d_in[1];   // [B, 8]
    const int*   cmd  = (const int*)  d_in[2];   // [B]
    const float* Wm   = (const float*)d_in[3];   // [8, 512]
    const float* bmv  = (const float*)d_in[4];   // [512]
    const float* W1   = (const float*)d_in[5];   // [4, 1024, 512]
    const float* b1   = (const float*)d_in[6];   // [4, 512]
    const float* W2   = (const float*)d_in[7];   // [4, 512, 2]
    const float* b2   = (const float*)d_in[8];   // [4, 2]
    float* out = (float*)d_out;                  // [2*B] = angle ++ speed

    // workspace layout (all offsets 256B-aligned); total ~4.3 MB
    char* ws = (char*)d_ws;
    _Float16* W1t   = (_Float16*)(ws);                           // 4 MB
    float*    w2col = (float*)   (ws + (size_t)4 * 1024 * 1024 + 256 * 1024);
    int*      perm  = (int*)     (ws + (size_t)4 * 1024 * 1024 + 272 * 1024);
    int*      meta  = (int*)     (ws + (size_t)4 * 1024 * 1024 + 352 * 1024);

    k_w1_transpose<<<(NCMD * EMB * K_DIM) / 256, 256, 0, stream>>>(W1, W1t);
    k_w2_col0     <<<(NCMD * EMB) / 256,        256, 0, stream>>>(W2, w2col);
    k_sort_init   <<<(PERM_LEN + 255) / 256,    256, 0, stream>>>(perm, meta);
    k_sort_hist   <<<B_ROWS / 256,              256, 0, stream>>>(cmd, meta);
    k_sort_seg    <<<1, 32, 0, stream>>>(meta);
    k_sort_scatter<<<B_ROWS / 256,              256, 0, stream>>>(cmd, meta, perm);
    k_moe_gemm    <<<MAX_TILES,                 256, 0, stream>>>(
        img, meas, Wm, bmv, b1, b2, W1t, w2col, perm, meta, out);
}
